// LSTMLayer_71279277245056
// MI455X (gfx1250) — compile-verified, matrix operand order checked by reading, not run on hardware
//
#include <hip/hip_runtime.h>

typedef __attribute__((ext_vector_type(16))) _Float16 v16h;
typedef __attribute__((ext_vector_type(8)))  _Float16 v8h;
typedef __attribute__((ext_vector_type(8)))  float    v8f;

#define BB 64
#define TT 256
#define DD 512
#define HH 512
#define LL 2
#define K1 1024      // D + H
#define N1 2048      // 4H
#define KT1 (K1/32)  // 32 k-tiles for gate GEMM
#define KT2 (HH/32)  // 16 k-tiles for keys GEMM
#define BS (BB*HH)   // one h/c state plane: 32768 elems

__device__ __forceinline__ v8f wmma_f16(v16h a, v16h b, v8f c) {
  return __builtin_amdgcn_wmma_f32_16x16x32_f16(
      /*neg_a=*/false, a, /*neg_b=*/false, b,
      /*c_mod=*/(short)0, c, /*reuse_a=*/false, /*reuse_b=*/false);
}

__device__ __forceinline__ float sigmoidf(float x) {
  return 1.0f / (1.0f + __expf(-x));
}

// Build a 16x32 f16 A-fragment per ISA layout:
// lanes 0-15 : row = lane,    halves 0-7 -> K kbase+0..7,  halves 8-15 -> K kbase+16..23
// lanes 16-31: row = lane-16, halves 0-7 -> K kbase+8..15, halves 8-15 -> K kbase+24..31
__device__ __forceinline__ v16h load_a_frag(const _Float16* __restrict__ rowptr) {
  v8h lo = *(const v8h*)(rowptr);
  v8h hi = *(const v8h*)(rowptr + 16);
  v16h a;
#pragma unroll
  for (int i = 0; i < 8; ++i) { a[i] = lo[i]; a[i + 8] = hi[i]; }
  return a;
}

// ---------------- conversion / packing -------------------------------------

__global__ void cvt_f32_to_f16(const float* __restrict__ src,
                               _Float16* __restrict__ dst, int n) {
  int i = blockIdx.x * blockDim.x + threadIdx.x;
  if (i < n) dst[i] = (_Float16)src[i];
}

// Pack row-major f32 weights [K][N] into WMMA B-fragment order:
// frag = nt*(K/32)+kt ; within frag: element index = lane*16 + half ;
// n = nt*16 + (lane&15) ; k = kt*32 + ((lane>=16)?8:0) + (half<8 ? half : half+8)
__global__ void pack_w_frag(const float* __restrict__ src,
                            _Float16* __restrict__ dst, int K, int N) {
  int e = blockIdx.x * blockDim.x + threadIdx.x;
  if (e >= K * N) return;
  int half = e & 15;
  int lane = (e >> 4) & 31;
  int frag = e >> 9;
  int KT = K >> 5;
  int kt = frag % KT;
  int nt = frag / KT;
  int base = (lane >> 4) << 3;
  int hh = (half < 8) ? half : (half + 8);
  int k = kt * 32 + base + hh;
  int n = nt * 16 + (lane & 15);
  dst[e] = (_Float16)src[(size_t)k * N + n];
}

__global__ void init_state(float* __restrict__ cst, _Float16* __restrict__ hb) {
  int i = blockIdx.x * blockDim.x + threadIdx.x;
  if (i < LL * BS) cst[i] = 0.0f;
  if (i < 2 * LL * BS) hb[i] = (_Float16)0.0f;
}

__global__ void pq_kernel(const float* __restrict__ q,
                          const float* __restrict__ Wq,
                          float* __restrict__ pq) {
  int u = blockIdx.x * blockDim.x + threadIdx.x;
  if (u >= HH) return;
  float acc = 0.0f;
  for (int k = 0; k < HH; ++k) acc += q[k] * Wq[(size_t)k * HH + u];
  pq[u] = acc;
}

// ---------------- recurrent LSTM step (one layer) --------------------------
// Z = [x_t | h_prev] @ W  (M=64, K=1024, N=2048), fused gates + state update.
// 32 waves (8 blocks x 4 waves, 1 wave/SIMD so the full VGPR file is ours):
// wave -> ut in 0..31. Each wave computes FULL M=64 (4 m-tiles) x 4 gate
// tiles, so every weight fragment is read from L2 exactly once per step.
// K loop split into x-half / h-half; streaming pointers, no per-iter selects.
__global__ __launch_bounds__(128, 1) void lstm_step_kernel(
    const _Float16* __restrict__ xbase, long xstride,
    const _Float16* __restrict__ hprev,
    const _Float16* __restrict__ wfrag,
    const float* __restrict__ bias,
    const int* __restrict__ lengths,
    float* __restrict__ cst,
    _Float16* __restrict__ hout,
    _Float16* __restrict__ yout,
    int writeY, int t) {
  int lane = threadIdx.x & 31;
  int ut = blockIdx.x * (blockDim.x >> 5) + (threadIdx.x >> 5); // 0..31

  v8f acc[4][4]; // [gate][mtile]
#pragma unroll
  for (int g = 0; g < 4; ++g)
#pragma unroll
    for (int mt = 0; mt < 4; ++mt) acc[g][mt] = v8f{};

  int rbase = lane & 15;
  int klane = (lane >> 4) << 3; // +0 or +8

  // B streams linearly: frag index (g*32+ut)*32 + kt, +512 elems per kt.
  const _Float16* bptr[4];
#pragma unroll
  for (int g = 0; g < 4; ++g)
    bptr[g] = wfrag + (((size_t)(g * 32 + ut) * KT1) << 9) + lane * 16;

  const _Float16* aptr[4];
#pragma unroll
  for (int mt = 0; mt < 4; ++mt)
    aptr[mt] = xbase + (size_t)(mt * 16 + rbase) * xstride + klane;

  // ---- x half: k = 0..511 ----
  for (int kt = 0; kt < 16; ++kt) {
    v16h a[4];
#pragma unroll
    for (int mt = 0; mt < 4; ++mt) { a[mt] = load_a_frag(aptr[mt]); aptr[mt] += 32; }
#pragma unroll
    for (int g = 0; g < 4; ++g) {
      v16h b = *(const v16h*)bptr[g];
      bptr[g] += 512;
#pragma unroll
      for (int mt = 0; mt < 4; ++mt) acc[g][mt] = wmma_f16(a[mt], b, acc[g][mt]);
    }
  }

  // ---- h half: k = 512..1023 ----
#pragma unroll
  for (int mt = 0; mt < 4; ++mt)
    aptr[mt] = hprev + (size_t)(mt * 16 + rbase) * HH + klane;
  for (int kt = 0; kt < 16; ++kt) {
    v16h a[4];
#pragma unroll
    for (int mt = 0; mt < 4; ++mt) { a[mt] = load_a_frag(aptr[mt]); aptr[mt] += 32; }
#pragma unroll
    for (int g = 0; g < 4; ++g) {
      v16h b = *(const v16h*)bptr[g];
      bptr[g] += 512;
#pragma unroll
      for (int mt = 0; mt < 4; ++mt) acc[g][mt] = wmma_f16(a[mt], b, acc[g][mt]);
    }
  }

  // epilogue: gates i,j,f,o  (columns u, u+512, u+1024, u+1536)
  int u = ut * 16 + (lane & 15);
  float bi = bias[u];
  float bj = bias[HH + u];
  float bf = bias[2 * HH + u];
  float bo = bias[3 * HH + u];
#pragma unroll
  for (int mt = 0; mt < 4; ++mt) {
#pragma unroll
    for (int r = 0; r < 8; ++r) {
      int row = mt * 16 + r + ((lane >> 4) << 3); // batch index
      int active = (t < lengths[row]) ? 1 : 0;
      float iv = acc[0][mt][r] + bi;
      float jv = acc[1][mt][r] + bj;
      float fv = acc[2][mt][r] + bf + 1.0f; // TF forget bias
      float ov = acc[3][mt][r] + bo;
      size_t idx = (size_t)row * HH + u;
      float c_old = cst[idx];
      float c_new = c_old * sigmoidf(fv) + sigmoidf(iv) * tanhf(jv);
      float h_new = sigmoidf(ov) * tanhf(c_new);
      float h_old = (float)hprev[idx];
      float hv = active ? h_new : h_old;
      if (active) cst[idx] = c_new;
      hout[idx] = (_Float16)hv;
      if (writeY) {
        float y = active ? h_new : 0.0f;
        yout[((size_t)row * TT + t) * HH + u] = (_Float16)y;
      }
    }
  }
}

// ---------------- attention score: v . tanh(Y@Wm + pq) ---------------------
// Rows = B*T = 16384. Each wave owns 4 row-tiles (64 rows) and loops over all
// 32 u-tiles (K=512), reusing each Wm fragment across 4 WMMAs. 256 waves.
__global__ __launch_bounds__(256, 1) void score_kernel(
    const _Float16* __restrict__ yf16,
    const _Float16* __restrict__ wmfrag,
    const float* __restrict__ pq,
    const float* __restrict__ vvec,
    float* __restrict__ score) {
  int lane = threadIdx.x & 31;
  int grp = blockIdx.x * (blockDim.x >> 5) + (threadIdx.x >> 5); // 0..255
  int row0 = grp * 64;
  int rbase = lane & 15;
  int klane = (lane >> 4) << 3;

  float sLocal[4][8];
#pragma unroll
  for (int mt = 0; mt < 4; ++mt)
#pragma unroll
    for (int r = 0; r < 8; ++r) sLocal[mt][r] = 0.0f;

  // B pointer walks the whole fragment buffer linearly (frag = ut*16 + kt).
  const _Float16* bp = wmfrag + lane * 16;

  for (int ut = 0; ut < 32; ++ut) {
    v8f acc[4];
#pragma unroll
    for (int mt = 0; mt < 4; ++mt) acc[mt] = v8f{};
    const _Float16* ap[4];
#pragma unroll
    for (int mt = 0; mt < 4; ++mt)
      ap[mt] = yf16 + (size_t)(row0 + mt * 16 + rbase) * HH + klane;
    for (int kt = 0; kt < KT2; ++kt) {
      v16h b = *(const v16h*)bp;
      bp += 512;
#pragma unroll
      for (int mt = 0; mt < 4; ++mt) {
        v16h a = load_a_frag(ap[mt]);
        ap[mt] += 32;
        acc[mt] = wmma_f16(a, b, acc[mt]);
      }
    }
    int u = ut * 16 + (lane & 15);
    float pqu = pq[u];
    float vu = vvec[u];
#pragma unroll
    for (int mt = 0; mt < 4; ++mt)
#pragma unroll
      for (int r = 0; r < 8; ++r) sLocal[mt][r] += tanhf(acc[mt][r] + pqu) * vu;
  }

#pragma unroll
  for (int mt = 0; mt < 4; ++mt) {
#pragma unroll
    for (int r = 0; r < 8; ++r) {
      float s = sLocal[mt][r];
      s += __shfl_xor(s, 1, 32);
      s += __shfl_xor(s, 2, 32);
      s += __shfl_xor(s, 4, 32);
      s += __shfl_xor(s, 8, 32);
      if ((lane & 15) == 0) {
        int row = row0 + mt * 16 + r + ((lane >> 4) << 3);
        score[row] = s;
      }
    }
  }
}

// ---------------- masked softmax over T per batch --------------------------
__global__ void softmax_kernel(const float* __restrict__ score,
                               const int* __restrict__ lengths,
                               float* __restrict__ align) {
  __shared__ float red[TT];
  int b = blockIdx.x, t = threadIdx.x;
  int len = lengths[b];
  float s = (t < len) ? score[b * TT + t] : -3.4e38f;
  red[t] = s;
  __syncthreads();
  for (int off = TT / 2; off > 0; off >>= 1) {
    if (t < off) red[t] = fmaxf(red[t], red[t + off]);
    __syncthreads();
  }
  float mx = red[0];
  __syncthreads();
  float e = (t < len) ? __expf(s - mx) : 0.0f;
  red[t] = e;
  __syncthreads();
  for (int off = TT / 2; off > 0; off >>= 1) {
    if (t < off) red[t] += red[t + off];
    __syncthreads();
  }
  align[b * TT + t] = e / red[0];
}

// ---------------- context = sum_t align[b,t] * y[b,t,:] --------------------
__global__ void context_kernel(const float* __restrict__ align,
                               const _Float16* __restrict__ yf16,
                               float* __restrict__ out) {
  int tid = blockIdx.x * blockDim.x + threadIdx.x; // 0..32767
  int b = tid >> 9;
  int u = tid & 511;
  const _Float16* yp = yf16 + (size_t)b * TT * HH + u;
  const float* ap = align + b * TT;
  float acc = 0.0f;
  for (int t = 0; t < TT; ++t) acc += ap[t] * (float)yp[(size_t)t * HH];
  out[tid] = acc;
}

// ---------------- launch ----------------------------------------------------

extern "C" void kernel_launch(void* const* d_in, const int* in_sizes, int n_in,
                              void* d_out, int out_size, void* d_ws, size_t ws_size,
                              hipStream_t stream) {
  const float* inputs  = (const float*)d_in[0]; // [B,T,D]
  const int*   lengths = (const int*)  d_in[1]; // [B]
  const float* Ws      = (const float*)d_in[2]; // [L, D+H, 4H]
  const float* bs      = (const float*)d_in[3]; // [L, 4H]
  const float* Wm      = (const float*)d_in[4]; // [H,H]
  const float* Wq      = (const float*)d_in[5]; // [H,H]
  const float* vvec    = (const float*)d_in[6]; // [H]
  const float* q       = (const float*)d_in[7]; // [1,H]
  float* out = (float*)d_out;                    // [B,H]

  // workspace carve-up (all offsets multiples of 256 bytes)
  char* w = (char*)d_ws;
  size_t off = 0;
  _Float16* xf16   = (_Float16*)(w + off); off += (size_t)BB * TT * DD * 2;   // 16 MB
  _Float16* wfrag  = (_Float16*)(w + off); off += (size_t)LL * K1 * N1 * 2;   // 8 MB
  _Float16* wmfrag = (_Float16*)(w + off); off += (size_t)HH * HH * 2;        // 512 KB
  _Float16* yf16   = (_Float16*)(w + off); off += (size_t)BB * TT * HH * 2;   // 16 MB
  _Float16* hb     = (_Float16*)(w + off); off += (size_t)2 * LL * BS * 2;    // 256 KB
  float*    cst    = (float*)   (w + off); off += (size_t)LL * BS * 4;        // 256 KB
  float*    pq     = (float*)   (w + off); off += 2048;
  float*    score  = (float*)   (w + off); off += (size_t)BB * TT * 4;
  float*    align  = (float*)   (w + off); off += (size_t)BB * TT * 4;
  (void)ws_size; (void)in_sizes; (void)n_in; (void)out_size;

  // one-time (per call) conversions and packing
  int nx = BB * TT * DD;
  cvt_f32_to_f16<<<(nx + 255) / 256, 256, 0, stream>>>(inputs, xf16, nx);
  for (int l = 0; l < LL; ++l) {
    pack_w_frag<<<(K1 * N1 + 255) / 256, 256, 0, stream>>>(
        Ws + (size_t)l * K1 * N1, wfrag + (size_t)l * K1 * N1, K1, N1);
  }
  pack_w_frag<<<(HH * HH + 255) / 256, 256, 0, stream>>>(Wm, wmfrag, HH, HH);
  pq_kernel<<<(HH + 255) / 256, 256, 0, stream>>>(q, Wq, pq);
  init_state<<<(2 * LL * BS + 255) / 256, 256, 0, stream>>>(cst, hb);

  _Float16* hb0 = hb;            // layer 0: two ping-pong planes
  _Float16* hb1 = hb + 2 * BS;   // layer 1

  for (int t = 0; t < TT; ++t) {
    int pp = t & 1, pn = (t + 1) & 1;
    // layer 0: x part comes from converted inputs at time t
    lstm_step_kernel<<<8, 128, 0, stream>>>(
        xf16 + (size_t)t * DD, (long)TT * DD,
        hb0 + (size_t)pp * BS, wfrag, bs, lengths,
        cst, hb0 + (size_t)pn * BS, yf16, /*writeY=*/0, t);
    // layer 1: x part = layer-0 fresh h
    lstm_step_kernel<<<8, 128, 0, stream>>>(
        hb0 + (size_t)pn * BS, (long)HH,
        hb1 + (size_t)pp * BS, wfrag + (size_t)K1 * N1, bs + N1, lengths,
        cst + BS, hb1 + (size_t)pn * BS, yf16, /*writeY=*/1, t);
  }

  score_kernel<<<32, 256, 0, stream>>>(yf16, wmfrag, pq, vvec, score);
  softmax_kernel<<<BB, TT, 0, stream>>>(score, lengths, align);
  context_kernel<<<BB * HH / 256, 256, 0, stream>>>(align, yf16, out);
}
